// Model_79147657330979
// MI455X (gfx1250) — compile-verified, matrix operand order checked by reading, not run on hardware
//
#include <hip/hip_runtime.h>
#include <hip/hip_bf16.h>

#define N_NODES 100000
#define N_EDGES 1600000
#define F_HID   128
#define F_OUT   64

typedef __attribute__((ext_vector_type(16))) __bf16 v16bf;
typedef __attribute__((ext_vector_type(8)))  float  v8f;

// ---------------------------------------------------------------- degree/norm
__global__ void k_init_deg(float* deg, int n) {
    int i = blockIdx.x * blockDim.x + threadIdx.x;
    if (i < n) deg[i] = 1.0f;                        // self-loop contribution
}

__global__ void k_deg_accum(const long long* __restrict__ ei, float* deg, int e) {
    int i = blockIdx.x * blockDim.x + threadIdx.x;
    if (i < e) atomicAdd(&deg[(int)ei[(size_t)N_EDGES + i]], 1.0f);
}

__global__ void k_dis(const float* __restrict__ deg, float* __restrict__ dis, int n) {
    int i = blockIdx.x * blockDim.x + threadIdx.x;
    if (i < n) {
        float d = deg[i];
        dis[i] = (d > 0.0f) ? rsqrtf(fmaxf(d, 1e-12f)) : 0.0f;
    }
}

// ---------------------------------------------------------------- WMMA GEMM
// C[nrows x NCOLS] = A[nrows x 128] @ W[128 x NCOLS] (+ b1 + b2)
// One wave computes a 16-row x NCOLS tile; block = 8 waves = 128 rows.
// Weights converted fp32->bf16 and transposed into LDS once per block.
template <int NCOLS>
__global__ __launch_bounds__(256)
void k_gemm_bf16(const float* __restrict__ A, const float* __restrict__ W,
                 const float* __restrict__ b1, const float* __restrict__ b2,
                 float* __restrict__ C, int nrows) {
    constexpr int K    = 128;
    constexpr int LDSW = K + 8;                      // pad 8 bf16 -> conflict-free b128 reads
    constexpr int NT   = NCOLS / 16;
    __shared__ __bf16 s_wt[NCOLS * LDSW];

    // cooperative transpose + convert: s_wt[n][k] = bf16(W[k][n])
    for (int i = threadIdx.x; i < K * NCOLS; i += 256) {
        int k = i / NCOLS, n = i % NCOLS;
        s_wt[n * LDSW + k] = (__bf16)W[i];
    }
    __syncthreads();

    const int lane = threadIdx.x & 31;
    const int wave = threadIdx.x >> 5;
    const int base = (lane >= 16) ? 8 : 0;           // K-half selector per WMMA A/B layout
    const int nlo  = lane & 15;
    const int row0 = blockIdx.x * 128 + wave * 16;
    const int mrow = row0 + nlo;
    const int arow_idx = (mrow < nrows) ? mrow : (nrows - 1);  // clamp: keep EXEC all-1s

    v8f acc[NT] = {};

    #pragma unroll
    for (int kk = 0; kk < K; kk += 32) {
        const float* ar = A + (size_t)arow_idx * K + kk + base;
        v16bf a;
        #pragma unroll
        for (int j = 0; j < 8; ++j) {
            a[j]     = (__bf16)ar[j];                // VGPR0..3 : K = base+0..7
            a[8 + j] = (__bf16)ar[16 + j];           // VGPR4..7 : K = 16+base+0..7
        }
        #pragma unroll
        for (int t = 0; t < NT; ++t) {
            const __bf16* wr = &s_wt[(t * 16 + nlo) * LDSW + kk + base];
            v16bf b;
            #pragma unroll
            for (int j = 0; j < 8; ++j) {
                b[j]     = wr[j];
                b[8 + j] = wr[16 + j];
            }
            acc[t] = __builtin_amdgcn_wmma_f32_16x16x32_bf16(
                false, a, false, b, (short)0, acc[t], false, false);
        }
    }

    // epilogue: C/D layout -> lane holds col = nlo, rows = rbase + r (r = VGPR idx)
    const int rbase = row0 + ((lane >= 16) ? 8 : 0);
    #pragma unroll
    for (int t = 0; t < NT; ++t) {
        int col = t * 16 + nlo;
        float bias = (b1 ? b1[col] : 0.0f) + (b2 ? b2[col] : 0.0f);
        #pragma unroll
        for (int r = 0; r < 8; ++r) {
            int row = rbase + r;
            if (row < nrows) C[(size_t)row * NCOLS + col] = acc[t][r] + bias;
        }
    }
}

// ---------------------------------------------------------------- edge scatter
// item < E: real edge (s=ei[0][e], d=ei[1][e]); item >= E: self-loop node.
// One wave per item: 32 lanes x float4 = 128 features, fp32 global atomics.
__global__ __launch_bounds__(256)
void k_scatter(const long long* __restrict__ ei, const float* __restrict__ dis,
               const float* __restrict__ P, float* __restrict__ H) {
    const int item = blockIdx.x * (blockDim.x >> 5) + (threadIdx.x >> 5);
    const int lane = threadIdx.x & 31;
    if (item >= N_EDGES + N_NODES) return;

    int s, d;
    if (item < N_EDGES) {
        s = (int)ei[item];
        d = (int)ei[(size_t)N_EDGES + item];
    } else {
        s = d = item - N_EDGES;
    }
    const float w = dis[s] * dis[d];

    const float4 v = *reinterpret_cast<const float4*>(P + (size_t)s * F_HID + lane * 4);
    float* hp = H + (size_t)d * F_HID + lane * 4;
    atomicAdd(hp + 0, v.x * w);
    atomicAdd(hp + 1, v.y * w);
    atomicAdd(hp + 2, v.z * w);
    atomicAdd(hp + 3, v.w * w);
}

// ---------------------------------------------------------------- leaky relu
__global__ void k_lrelu(float* __restrict__ x, size_t n) {
    size_t i = (size_t)blockIdx.x * blockDim.x + threadIdx.x;
    if (i < n) {
        float v = x[i];
        x[i] = (v > 0.0f) ? v : 0.01f * v;
    }
}

// ---------------------------------------------------------------- launcher
extern "C" void kernel_launch(void* const* d_in, const int* in_sizes, int n_in,
                              void* d_out, int out_size, void* d_ws, size_t ws_size,
                              hipStream_t stream) {
    const float*     X   = (const float*)d_in[0];
    const long long* ei  = (const long long*)d_in[1];
    const float* gw1 = (const float*)d_in[3];
    const float* gb1 = (const float*)d_in[4];
    const float* lw1 = (const float*)d_in[5];
    const float* lb1 = (const float*)d_in[6];
    const float* gw2 = (const float*)d_in[7];
    const float* gb2 = (const float*)d_in[8];
    const float* lw2 = (const float*)d_in[9];
    const float* lb2 = (const float*)d_in[10];
    const float* gw3 = (const float*)d_in[11];
    const float* gb3 = (const float*)d_in[12];
    const float* lw3 = (const float*)d_in[13];
    const float* lb3 = (const float*)d_in[14];
    const float* fcw = (const float*)d_in[15];
    const float* fcb = (const float*)d_in[16];

    float* ws   = (float*)d_ws;
    float* deg  = ws;
    float* dis  = ws + N_NODES;
    float* bufA = ws + 2 * (size_t)N_NODES;
    float* bufB = bufA + (size_t)N_NODES * F_HID;
    float* P    = bufB + (size_t)N_NODES * F_HID;

    float* h_out = (float*)d_out;                           // [N, 128]
    float* y_out = h_out + (size_t)N_NODES * F_HID;         // [N, 64]

    const dim3 b256(256);
    const int gblocks = (N_NODES + 127) / 128;
    const int sblocks = (N_EDGES + N_NODES + 7) / 8;
    const size_t nh   = (size_t)N_NODES * F_HID;
    const int eblocks = (int)((nh + 255) / 256);

    // normalization weights
    k_init_deg <<<(N_NODES + 255) / 256, b256, 0, stream>>>(deg, N_NODES);
    k_deg_accum<<<(N_EDGES + 255) / 256, b256, 0, stream>>>(ei, deg, N_EDGES);
    k_dis      <<<(N_NODES + 255) / 256, b256, 0, stream>>>(deg, dis, N_NODES);

    // layer 1: bufA = lrelu( scatter(X@gw1) + X@lw1 + lb1 + gb1 )
    k_gemm_bf16<128><<<gblocks, b256, 0, stream>>>(X, gw1, nullptr, nullptr, P, N_NODES);
    k_gemm_bf16<128><<<gblocks, b256, 0, stream>>>(X, lw1, lb1, gb1, bufA, N_NODES);
    k_scatter<<<sblocks, b256, 0, stream>>>(ei, dis, P, bufA);
    k_lrelu  <<<eblocks, b256, 0, stream>>>(bufA, nh);

    // layer 2
    k_gemm_bf16<128><<<gblocks, b256, 0, stream>>>(bufA, gw2, nullptr, nullptr, P, N_NODES);
    k_gemm_bf16<128><<<gblocks, b256, 0, stream>>>(bufA, lw2, lb2, gb2, bufB, N_NODES);
    k_scatter<<<sblocks, b256, 0, stream>>>(ei, dis, P, bufB);
    k_lrelu  <<<eblocks, b256, 0, stream>>>(bufB, nh);

    // layer 3 -> h region of d_out
    k_gemm_bf16<128><<<gblocks, b256, 0, stream>>>(bufB, gw3, nullptr, nullptr, P, N_NODES);
    k_gemm_bf16<128><<<gblocks, b256, 0, stream>>>(bufB, lw3, lb3, gb3, h_out, N_NODES);
    k_scatter<<<sblocks, b256, 0, stream>>>(ei, dis, P, h_out);
    k_lrelu  <<<eblocks, b256, 0, stream>>>(h_out, nh);

    // final linear: y = h @ fcw + fcb
    k_gemm_bf16<64><<<gblocks, b256, 0, stream>>>(h_out, fcw, fcb, nullptr, y_out, N_NODES);
}
